// HybridFusionNet_76459007803748
// MI455X (gfx1250) — compile-verified, matrix operand order checked by reading, not run on hardware
//
#include <hip/hip_runtime.h>
#include <hip/hip_bf16.h>
#include <math.h>

typedef __attribute__((ext_vector_type(16))) _Float16 v16h;
typedef __attribute__((ext_vector_type(8)))  _Float16 v8h;
typedef __attribute__((ext_vector_type(8)))  float    v8f;

#define HW512 (512*512)
#define HP 256
#define LTOK (2*HP*HP)          // 131072 tokens total (B=2)
#define LN_EPS 1e-5f

// ---------------------------------------------------------------------------
// Generic WMMA GEMM: C[M,N] = act(bn(A*Bw + bias)) (+resid), f16 in, f32 acc.
// Block tile 256x64: 8 waves, each wave owns two 16-row strips x 64 cols.
// AMODE 0: row-major f16 [M,K].
// AMODE 1: implicit im2col from NHWC f16 (Cin channels), k=(ky*KW+kx)*Cin+ci.
// AMODE 2: like 1, but channels gathered from up to three 64-ch NHWC tensors
//          (A, A2, A3) concatenated along C (Cin = 64*nsrc).
// Bw packed in per-lane WMMA B-fragment order (see pack_wfrag).
// smode 0: row-major [M,N] output (== NHWC for convs).
// smode 1: conv-transpose scatter (row=b*65536+py*256+px, col=co*4+dy*2+dx)
//          -> NHWC [2,512,512,64] f16.
// Assumes M%256==0, K%32==0, N%16==0, Cin%32==0.
// ---------------------------------------------------------------------------
struct GemmP {
    const _Float16* A;
    const _Float16* A2;
    const _Float16* A3;
    const _Float16* Bw;      // fragment-packed
    const float* bias;       // [N] or null
    const float* bnS;        // [N] scale or null
    const float* bnB;        // [N] shift
    const float* resid;      // [M,N] f32 or null
    float*      out32;       // optional
    _Float16*   out16;       // optional
    int M, N, K;
    int amode;
    int act;                 // 0 none, 1 relu, 2 gelu(exact)
    int nsc;                 // cols [0,nsc) get *sval in epilogue (q-scale)
    float sval;
    int nbase0;              // column offset of this launch
    int smode;               // 0 rowmajor, 1 convT scatter
    int Cin, Hin, Win, KH, KW, stride, pad, Hout, Wout;
};

template <int NT, int AMODE>
__global__ __launch_bounds__(256) void gemm_wmma(GemmP p) {
    const int wave = threadIdx.x >> 5;
    const int lane = threadIdx.x & 31;
    const int lh = lane & 15, hi = lane >> 4;
    const int m0 = blockIdx.x * 256 + wave * 32;
    const int nbase = p.nbase0 + blockIdx.y * 64;
    const int ntiles = p.N >> 4;
    const int ntG0 = nbase >> 4;
    const v16h* Bf = (const v16h*)p.Bw;
    const int arow0 = m0 + lh;
    const int arow1 = arow0 + 16;

    v8f acc[2][NT];
#pragma unroll
    for (int s = 0; s < 2; ++s)
#pragma unroll
        for (int nt = 0; nt < NT; ++nt) acc[s][nt] = (v8f){};

    auto mma_step = [&](v8h a00, v8h a01, v8h a10, v8h a11, int kb) {
        v16h a0, a1;
#pragma unroll
        for (int j = 0; j < 8; ++j) {
            a0[j] = a00[j]; a0[8 + j] = a01[j];
            a1[j] = a10[j]; a1[8 + j] = a11[j];
        }
        v16h bfr[NT];
#pragma unroll
        for (int nt = 0; nt < NT; ++nt)
            bfr[nt] = Bf[((long)kb * ntiles + ntG0 + nt) * 32 + lane];
#pragma unroll
        for (int nt = 0; nt < NT; ++nt)
            acc[0][nt] = __builtin_amdgcn_wmma_f32_16x16x32_f16(
                false, a0, false, bfr[nt], (short)0, acc[0][nt], false, false);
#pragma unroll
        for (int nt = 0; nt < NT; ++nt)
            acc[1][nt] = __builtin_amdgcn_wmma_f32_16x16x32_f16(
                false, a1, false, bfr[nt], (short)0, acc[1][nt], false, false);
    };

    if (AMODE == 0) {
        const _Float16* ab0 = p.A + (long)arow0 * p.K;
        const _Float16* ab1 = p.A + (long)arow1 * p.K;
        for (int k0 = 0; k0 < p.K; k0 += 32) {
            const v8h a00 = *(const v8h*)(ab0 + k0 + hi * 8);
            const v8h a01 = *(const v8h*)(ab0 + k0 + 16 + hi * 8);
            const v8h a10 = *(const v8h*)(ab1 + k0 + hi * 8);
            const v8h a11 = *(const v8h*)(ab1 + k0 + 16 + hi * 8);
            mma_step(a00, a01, a10, a11, k0 >> 5);
        }
    } else {
        const int hw = p.Hout * p.Wout;
        const int b0 = arow0 / hw, r0 = arow0 - b0 * hw;
        const int oy0 = r0 / p.Wout, ox0 = r0 - oy0 * p.Wout;
        const int b1 = arow1 / hw, r1 = arow1 - b1 * hw;
        const int oy1 = r1 / p.Wout, ox1 = r1 - oy1 * p.Wout;
        int kb = 0;
        for (int ky = 0; ky < p.KH; ++ky) {
            const int iy0 = oy0 * p.stride - p.pad + ky;
            const int iy1 = oy1 * p.stride - p.pad + ky;
            for (int kx = 0; kx < p.KW; ++kx) {
                const int ix0 = ox0 * p.stride - p.pad + kx;
                const int ix1 = ox1 * p.stride - p.pad + kx;
                const bool v0 = ((unsigned)iy0 < (unsigned)p.Hin) &&
                                ((unsigned)ix0 < (unsigned)p.Win);
                const bool v1 = ((unsigned)iy1 < (unsigned)p.Hin) &&
                                ((unsigned)ix1 < (unsigned)p.Win);
                if (AMODE == 1) {
                    const _Float16* ap0 =
                        p.A + (((long)b0 * p.Hin + iy0) * p.Win + ix0) * p.Cin;
                    const _Float16* ap1 =
                        p.A + (((long)b1 * p.Hin + iy1) * p.Win + ix1) * p.Cin;
                    for (int ci0 = 0; ci0 < p.Cin; ci0 += 32, ++kb) {
                        v8h a00 = {}, a01 = {}, a10 = {}, a11 = {};
                        if (v0) {
                            a00 = *(const v8h*)(ap0 + ci0 + hi * 8);
                            a01 = *(const v8h*)(ap0 + ci0 + 16 + hi * 8);
                        }
                        if (v1) {
                            a10 = *(const v8h*)(ap1 + ci0 + hi * 8);
                            a11 = *(const v8h*)(ap1 + ci0 + 16 + hi * 8);
                        }
                        mma_step(a00, a01, a10, a11, kb);
                    }
                } else {   // AMODE 2: multi-source, each 64-channel NHWC
                    const long pix0 =
                        (((long)b0 * p.Hin + iy0) * p.Win + ix0) * 64;
                    const long pix1 =
                        (((long)b1 * p.Hin + iy1) * p.Win + ix1) * 64;
                    for (int ci0 = 0; ci0 < p.Cin; ci0 += 32, ++kb) {
                        const int seg = ci0 >> 6;
                        const _Float16* src =
                            seg == 0 ? p.A : (seg == 1 ? p.A2 : p.A3);
                        const int cl = ci0 & 63;
                        v8h a00 = {}, a01 = {}, a10 = {}, a11 = {};
                        if (v0) {
                            a00 = *(const v8h*)(src + pix0 + cl + hi * 8);
                            a01 = *(const v8h*)(src + pix0 + cl + 16 + hi * 8);
                        }
                        if (v1) {
                            a10 = *(const v8h*)(src + pix1 + cl + hi * 8);
                            a11 = *(const v8h*)(src + pix1 + cl + 16 + hi * 8);
                        }
                        mma_step(a00, a01, a10, a11, kb);
                    }
                }
            }
        }
    }

    // Epilogue. C layout: VGPR r -> row base + hi*8 + r, col n0 + lh.
#pragma unroll
    for (int s = 0; s < 2; ++s) {
#pragma unroll
        for (int nt = 0; nt < NT; ++nt) {
            const int col = nbase + nt * 16 + lh;
#pragma unroll
            for (int r = 0; r < 8; ++r) {
                const int row = m0 + s * 16 + hi * 8 + r;
                float v = acc[s][nt][r];
                if (p.bias) v += p.bias[col];
                if (col < p.nsc) v *= p.sval;
                if (p.bnS)  v = v * p.bnS[col] + p.bnB[col];
                if (p.act == 1)      v = fmaxf(v, 0.f);
                else if (p.act == 2) v = 0.5f * v * (1.f + erff(v * 0.70710678118f));
                if (p.resid) v += p.resid[(long)row * p.N + col];
                long oidx;
                if (p.smode == 1) {
                    const int bb = row >> 16;           // row = b*65536 + pix
                    const int pix = row & 65535;
                    const int py = pix >> 8, px = pix & 255;
                    const int co = col >> 2;
                    const int dy = (col >> 1) & 1, dx = col & 1;
                    const long opix =
                        ((long)bb * 512 + 2 * py + dy) * 512 + 2 * px + dx;
                    oidx = opix * 64 + co;
                } else {
                    oidx = (long)row * p.N + col;
                }
                if (p.out32) p.out32[oidx] = v;
                if (p.out16) p.out16[oidx] = (_Float16)v;
            }
        }
    }
}

// ---------------------------------------------------------------------------
// Pack weights into WMMA B-fragment order.
// Dest index: (((k/32)*(N/16) + n/16)*32 + lane)*16 + e, with
//   lh=lane&15, hi=lane>>4, e<8 -> k=kb*32+hi*8+e, e>=8 -> k=kb*32+16+hi*8+e-8,
//   n = nt*16+lh.
// srcRM=0: OIHW [N][Cin][KH][KW], k=(ky*KW+kx)*Cin+ci ;  srcRM=1: [K][N] RM.
// ---------------------------------------------------------------------------
__global__ void pack_wfrag(const float* __restrict__ W, _Float16* __restrict__ Bp,
                           int N, int Cin, int KH, int KW, int srcRM) {
    const int K = Cin * KH * KW;
    const long tot = (long)K * N;
    const long i = (long)blockIdx.x * blockDim.x + threadIdx.x;
    if (i >= tot) return;
    const int e = (int)(i & 15);
    const int lane = (int)((i >> 4) & 31);
    const long rest = i >> 9;
    const int ntiles = N >> 4;
    const int nt = (int)(rest % ntiles);
    const int kb = (int)(rest / ntiles);
    const int lh = lane & 15, hi = lane >> 4;
    const int k = kb * 32 + (e < 8 ? hi * 8 + e : 16 + hi * 8 + (e - 8));
    const int n = nt * 16 + lh;
    float v;
    if (srcRM) {
        v = W[(long)k * N + n];
    } else {
        const int g = k / Cin, ci = k - g * Cin;
        const int ky = g / KW, kx = g - ky * KW;
        v = W[(((long)n * Cin + ci) * KH + ky) * KW + kx];
    }
    Bp[i] = (_Float16)v;
}

// expand convT bias: bt4[n] = bt[n>>2], n in [0,256)
__global__ void expand_bt(const float* __restrict__ bt, float* __restrict__ bt4) {
    const int i = threadIdx.x;
    bt4[i] = bt[i >> 2];
}

// ---------------------------------------------------------------------------
// Window attention: 1 block per (batch,window), 1 wave per head (4 waves).
// qkv is f16 [LTOK,288], q already scaled (folded into QKV GEMM epilogue).
// ---------------------------------------------------------------------------
__global__ __launch_bounds__(128) void attn_wmma(
    const _Float16* __restrict__ qkv,   // [LTOK, 288] f16 (q pre-scaled)
    _Float16* __restrict__ out16,       // [LTOK, 96]
    const float* __restrict__ rel_table,// [225, 4]
    int shift) {
    __shared__ float    S[4][64 * 64];   // 64 KB
    __shared__ _Float16 P16[4][64 * 64]; // 32 KB
    __shared__ _Float16 Vt[4][32 * 64];  // 16 KB (dims padded 24->32 with zeros)
    __shared__ float    BiasT[4][228];   // 3.6 KB rel-pos bias, head-major

    const int head = threadIdx.x >> 5;
    const int lane = threadIdx.x & 31;
    const int lh = lane & 15, hi = lane >> 4;
    const int win = blockIdx.x;
    const int b  = win >> 10;
    const int w  = win & 1023;
    const int wy = w >> 5, wx = w & 31;

    // stage bias table transposed: BiasT[h][idx] = rel_table[idx][h]
    for (int i = threadIdx.x; i < 225 * 4; i += 128)
        BiasT[i & 3][i >> 2] = rel_table[i];
    __syncthreads();

    auto trow = [&](int t) -> long {     // token -> global [B*L] row (shift-rolled)
        const int ty = t >> 3, tx = t & 7;
        const int y = (wy * 8 + ty + shift) & 255;
        const int x = (wx * 8 + tx + shift) & 255;
        return (long)b * (HP * HP) + y * HP + x;
    };

    float* Sh = S[head];
    _Float16* Ph = P16[head];
    _Float16* Vh = Vt[head];
    const float* Bh = BiasT[head];

    // ---- stage V transposed (+zero pad): Vt[d][t] = v[t][d] -------------
    for (int t = lane; t < 64; t += 32) {
        const _Float16* vp = qkv + trow(t) * 288 + 192 + head * 24;
#pragma unroll
        for (int d = 0; d < 24; ++d) Vh[d * 64 + t] = vp[d];
#pragma unroll
        for (int d = 24; d < 32; ++d) Vh[d * 64 + t] = (_Float16)0.f;
    }

    // ---- S = q k^T via 4x4 tiles of 16x16, K=32 (24 used, rest 0) -------
    for (int mi = 0; mi < 4; ++mi) {
        const _Float16* qp = qkv + trow(mi * 16 + lh) * 288 + head * 24;
        const v8h q0 = *(const v8h*)(qp + hi * 8);
        const v8h q1 = *(const v8h*)(qp + 16 + hi * 8);   // hi==1 part masked below
        v16h aq;
#pragma unroll
        for (int j = 0; j < 8; ++j) {
            aq[j] = q0[j];
            aq[8 + j] = hi ? (_Float16)0.f : q1[j];
        }
        for (int ni = 0; ni < 4; ++ni) {
            const _Float16* kp = qkv + trow(ni * 16 + lh) * 288 + 96 + head * 24;
            const v8h k0c = *(const v8h*)(kp + hi * 8);
            const v8h k1c = *(const v8h*)(kp + 16 + hi * 8);
            v16h bk;
#pragma unroll
            for (int j = 0; j < 8; ++j) {
                bk[j] = k0c[j];
                bk[8 + j] = hi ? (_Float16)0.f : k1c[j];
            }
            v8f c = {};
            c = __builtin_amdgcn_wmma_f32_16x16x32_f16(
                false, aq, false, bk, (short)0, c, false, false);
#pragma unroll
            for (int r = 0; r < 8; ++r)
                Sh[(mi * 16 + hi * 8 + r) * 64 + ni * 16 + lh] = c[r];
        }
    }

    // ---- bias + softmax (2 rows per lane) -> P16 ------------------------
    for (int rr = 0; rr < 2; ++rr) {
        const int m = lane * 2 + rr;
        const int my = m >> 3, mx = m & 7;
        float mmax = -1e30f;
        for (int n = 0; n < 64; ++n) {
            const int ny = n >> 3, nx = n & 7;
            const int idx = (my - ny + 7) * 15 + (mx - nx + 7);
            const float s = Sh[m * 64 + n] + Bh[idx];
            Sh[m * 64 + n] = s;
            mmax = fmaxf(mmax, s);
        }
        float sum = 0.f;
        for (int n = 0; n < 64; ++n) {
            const float e = expf(Sh[m * 64 + n] - mmax);
            Sh[m * 64 + n] = e;          // cache exp
            sum += e;
        }
        const float inv = 1.f / sum;
        for (int n = 0; n < 64; ++n)
            Ph[m * 64 + n] = (_Float16)(Sh[m * 64 + n] * inv);
    }

    // ---- O = P V : M=64 (4 tiles), N=24 (2 tiles), K=64 (2 chunks) ------
    for (int mi = 0; mi < 4; ++mi) {
        const int mrow = mi * 16 + lh;
        for (int nj = 0; nj < 2; ++nj) {
            const int ncol = nj * 16 + lh;
            v8f c = {};
            for (int kk = 0; kk < 2; ++kk) {
                const v8h a0 = *(const v8h*)(Ph + mrow * 64 + kk * 32 + hi * 8);
                const v8h a1 = *(const v8h*)(Ph + mrow * 64 + kk * 32 + 16 + hi * 8);
                const v8h b0 = *(const v8h*)(Vh + ncol * 64 + kk * 32 + hi * 8);
                const v8h b1 = *(const v8h*)(Vh + ncol * 64 + kk * 32 + 16 + hi * 8);
                v16h ap, bv;
#pragma unroll
                for (int j = 0; j < 8; ++j) {
                    ap[j] = a0[j]; ap[8 + j] = a1[j];
                    bv[j] = b0[j]; bv[8 + j] = b1[j];
                }
                c = __builtin_amdgcn_wmma_f32_16x16x32_f16(
                    false, ap, false, bv, (short)0, c, false, false);
            }
            if (ncol < 24) {
#pragma unroll
                for (int r = 0; r < 8; ++r)
                    out16[trow(mi * 16 + hi * 8 + r) * 96 + head * 24 + ncol] =
                        (_Float16)c[r];
            }
        }
    }
}

// ---------------------------------------------------------------------------
// Small kernels (activations in NHWC f16)
// ---------------------------------------------------------------------------
__global__ void conv1_direct(const float* __restrict__ img,
                             const float* __restrict__ w,
                             const float* __restrict__ bias,
                             const float* __restrict__ g, const float* __restrict__ bb,
                             const float* __restrict__ mean, const float* __restrict__ var,
                             _Float16* __restrict__ out) {   // NHWC [2,512,512,64]
    const long i = (long)blockIdx.x * blockDim.x + threadIdx.x;
    if (i >= (long)2 * 64 * HW512) return;
    const int co = (int)(i & 63);
    const long t = i >> 6;
    const int p = (int)(t % HW512);
    const int b = (int)(t / HW512);
    const int y = p >> 9, x = p & 511;
    float acc = bias[co];
    for (int ci = 0; ci < 3; ++ci)
        for (int ky = 0; ky < 3; ++ky) {
            const int iy = y - 1 + ky;
            if ((unsigned)iy >= 512u) continue;
            for (int kx = 0; kx < 3; ++kx) {
                const int ix = x - 1 + kx;
                if ((unsigned)ix >= 512u) continue;
                acc += img[((long)(b * 3 + ci) * 512 + iy) * 512 + ix] *
                       w[((co * 3 + ci) * 3 + ky) * 3 + kx];
            }
        }
    const float sc = g[co] * rsqrtf(var[co] + LN_EPS);
    const float v = fmaxf(acc * sc + (bb[co] - mean[co] * sc), 0.f);
    out[i] = (_Float16)v;
}

__global__ void ln_f16(const float* __restrict__ x, const float* __restrict__ g,
                       const float* __restrict__ bt, _Float16* __restrict__ out,
                       int ntok) {
    const int tok = blockIdx.x * 8 + (threadIdx.x >> 5);
    const int lane = threadIdx.x & 31;
    if (tok >= ntok) return;
    const float* xp = x + (long)tok * 96;
    const float v0 = xp[lane], v1 = xp[lane + 32], v2 = xp[lane + 64];
    float s = v0 + v1 + v2;
    for (int o = 16; o; o >>= 1) s += __shfl_xor(s, o, 32);
    const float mu = s * (1.f / 96.f);
    const float d0 = v0 - mu, d1 = v1 - mu, d2 = v2 - mu;
    float vs = d0 * d0 + d1 * d1 + d2 * d2;
    for (int o = 16; o; o >>= 1) vs += __shfl_xor(vs, o, 32);
    const float inv = rsqrtf(vs * (1.f / 96.f) + LN_EPS);
    _Float16* op = out + (long)tok * 96;
    op[lane]      = (_Float16)(d0 * inv * g[lane]      + bt[lane]);
    op[lane + 32] = (_Float16)(d1 * inv * g[lane + 32] + bt[lane + 32]);
    op[lane + 64] = (_Float16)(d2 * inv * g[lane + 64] + bt[lane + 64]);
}

__global__ void bn_prep(const float* g, const float* b, const float* mean,
                        const float* var, float* sc, float* sh, int C) {
    const int c = blockIdx.x * blockDim.x + threadIdx.x;
    if (c < C) {
        const float s = g[c] * rsqrtf(var[c] + LN_EPS);
        sc[c] = s;
        sh[c] = b[c] - mean[c] * s;
    }
}

__global__ void final_head(const _Float16* __restrict__ h,   // NHWC [2,512,512,64]
                           const float* __restrict__ w2, const float* __restrict__ b2,
                           const float* __restrict__ img1, const float* __restrict__ img2,
                           float* __restrict__ out) {
    const long i = (long)blockIdx.x * blockDim.x + threadIdx.x;
    if (i >= (long)2 * HW512) return;
    const int b = (int)(i / HW512);
    const int p = (int)(i % HW512);
    const _Float16* hp = h + i * 64;
    float acc = b2[0];
    for (int c = 0; c < 64; ++c) acc += (float)hp[c] * w2[c];
    float mk = 1.f / (1.f + expf(-acc));
    mk = fminf(fmaxf(mk, 1e-4f), 1.f - 1e-4f);
    float* fused = out;                           // [2,3,HW]
    float* maskp = out + (long)2 * 3 * HW512;     // [2,HW]
    float* logp  = maskp + (long)2 * HW512;       // [2,HW]
    for (int c = 0; c < 3; ++c) {
        const long ii = ((long)(b * 3 + c)) * HW512 + p;
        fused[ii] = mk * img1[ii] + (1.f - mk) * img2[ii];
    }
    maskp[i] = mk;
    logp[i]  = acc;
}

// ---------------------------------------------------------------------------
extern "C" void kernel_launch(void* const* d_in, const int* in_sizes, int n_in,
                              void* d_out, int out_size, void* d_ws, size_t ws_size,
                              hipStream_t stream) {
    auto in = [&](int i) { return (const float*)d_in[i]; };
    const float* img1 = in(0);
    const float* img2 = in(1);
    // enc: 2..13 ; patch: 14,15 ; blocks[0]: 16..28 ; blocks[1]: 29..41
    // norm: 42,43 ; up: 44..47 ; mask: 48..51
    const int B0 = 16, B1 = 29;

    uint8_t* ws = (uint8_t*)d_ws;
    size_t off = 0;
    auto alloc = [&](size_t bytes) -> void* {
        off = (off + 255) & ~(size_t)255;
        void* p = ws + off;
        off += bytes;
        return p;
    };

    const size_t featB = (size_t)2 * 64 * HW512 * sizeof(_Float16);      // 67 MB
    _Float16* e1a  = (_Float16*)alloc(featB);   // conv1(img1); later gT; later h
    _Float16* e1b  = (_Float16*)alloc(featB);   // conv1(img2); later g
    _Float16* f1h  = (_Float16*)alloc(featB);   // NHWC
    _Float16* f2h  = (_Float16*)alloc(featB);   // NHWC
    float*    x_a  = (float*)alloc((size_t)LTOK * 96 * sizeof(float));
    float*    x_b  = (float*)alloc((size_t)LTOK * 96 * sizeof(float));
    _Float16* tokA = (_Float16*)alloc((size_t)LTOK * 96 * sizeof(_Float16));
    _Float16* tokB = (_Float16*)alloc((size_t)LTOK * 96 * sizeof(_Float16));
    _Float16* scr16 = (_Float16*)alloc((size_t)LTOK * 384 * sizeof(_Float16));
    _Float16* qkv16 = scr16;                    // [LTOK,288]
    _Float16* mlp16 = scr16;                    // [LTOK,384]
    float*    bn2sc = (float*)alloc(64 * sizeof(float));
    float*    bn2sh = (float*)alloc(64 * sizeof(float));
    float*    bt4   = (float*)alloc(256 * sizeof(float));

    _Float16* w2p   = (_Float16*)alloc((size_t)576 * 64 * 2);
    _Float16* patp  = (_Float16*)alloc((size_t)512 * 96 * 2);
    _Float16* qkvp0 = (_Float16*)alloc((size_t)96 * 288 * 2);
    _Float16* qkvp1 = (_Float16*)alloc((size_t)96 * 288 * 2);
    _Float16* prjp0 = (_Float16*)alloc((size_t)96 * 96 * 2);
    _Float16* prjp1 = (_Float16*)alloc((size_t)96 * 96 * 2);
    _Float16* fc1p0 = (_Float16*)alloc((size_t)96 * 384 * 2);
    _Float16* fc1p1 = (_Float16*)alloc((size_t)96 * 384 * 2);
    _Float16* fc2p0 = (_Float16*)alloc((size_t)384 * 96 * 2);
    _Float16* fc2p1 = (_Float16*)alloc((size_t)384 * 96 * 2);
    _Float16* upwp  = (_Float16*)alloc((size_t)576 * 64 * 2);
    _Float16* mskp  = (_Float16*)alloc((size_t)1728 * 64 * 2);
    _Float16* ctwp  = (_Float16*)alloc((size_t)96 * 256 * 2);

    auto pack = [&](const float* W, _Float16* Bp, int N, int Cin, int KH, int KW,
                    int srcRM) {
        const long tot = (long)Cin * KH * KW * N;
        hipLaunchKernelGGL(pack_wfrag, dim3((unsigned)((tot + 255) / 256)), dim3(256),
                           0, stream, W, Bp, N, Cin, KH, KW, srcRM);
    };
    auto gemm = [&](GemmP p) {
        const int fullB = p.N >> 6;                 // # of 64-wide column blocks
        const int rem = (p.N & 63) >> 4;            // leftover 16-tiles (0 or 2 here)
        auto L = [&](int nt, int gy) {
            dim3 g(p.M / 256, gy), blk(256);
            if (p.amode == 0) {
                if (nt == 4) hipLaunchKernelGGL((gemm_wmma<4, 0>), g, blk, 0, stream, p);
                else         hipLaunchKernelGGL((gemm_wmma<2, 0>), g, blk, 0, stream, p);
            } else if (p.amode == 1) {
                if (nt == 4) hipLaunchKernelGGL((gemm_wmma<4, 1>), g, blk, 0, stream, p);
                else         hipLaunchKernelGGL((gemm_wmma<2, 1>), g, blk, 0, stream, p);
            } else {
                if (nt == 4) hipLaunchKernelGGL((gemm_wmma<4, 2>), g, blk, 0, stream, p);
                else         hipLaunchKernelGGL((gemm_wmma<2, 2>), g, blk, 0, stream, p);
            }
        };
        if (fullB) { p.nbase0 = 0;          L(4, fullB); }
        if (rem)   { p.nbase0 = fullB << 6; L(2, 1); }
    };
    auto ln = [&](const float* x, const float* g, const float* b, _Float16* o) {
        hipLaunchKernelGGL(ln_f16, dim3(LTOK / 8), dim3(256), 0, stream,
                           x, g, b, o, LTOK);
    };

    // ---- weight packing ----
    pack(in(8),       w2p,   64,  64,  3, 3, 0);
    pack(in(14),      patp,  96,  128, 2, 2, 0);
    pack(in(B0 + 3),  qkvp0, 288, 96,  1, 1, 0);
    pack(in(B1 + 3),  qkvp1, 288, 96,  1, 1, 0);
    pack(in(B0 + 5),  prjp0, 96,  96,  1, 1, 0);
    pack(in(B1 + 5),  prjp1, 96,  96,  1, 1, 0);
    pack(in(B0 + 9),  fc1p0, 384, 96,  1, 1, 0);
    pack(in(B1 + 9),  fc1p1, 384, 96,  1, 1, 0);
    pack(in(B0 + 11), fc2p0, 96,  384, 1, 1, 0);
    pack(in(B1 + 11), fc2p1, 96,  384, 1, 1, 0);
    pack(in(46),      upwp,  64,  64,  3, 3, 0);
    pack(in(48),      mskp,  64,  192, 3, 3, 0);
    pack(in(44),      ctwp,  256, 96,  1, 1, 1);   // wt already [K=96][N=256]
    hipLaunchKernelGGL(bn_prep, dim3(1), dim3(64), 0, stream,
                       in(10), in(11), in(12), in(13), bn2sc, bn2sh, 64);
    hipLaunchKernelGGL(expand_bt, dim3(1), dim3(256), 0, stream, in(45), bt4);

    // ---- encoder ----
    {
        const long tot = (long)2 * 64 * HW512;
        hipLaunchKernelGGL(conv1_direct, dim3((unsigned)((tot + 255) / 256)), dim3(256),
                           0, stream, img1, in(2), in(3), in(4), in(5), in(6), in(7), e1a);
        hipLaunchKernelGGL(conv1_direct, dim3((unsigned)((tot + 255) / 256)), dim3(256),
                           0, stream, img2, in(2), in(3), in(4), in(5), in(6), in(7), e1b);
    }
    GemmP c2{};
    c2.Bw = w2p; c2.bias = in(9); c2.bnS = bn2sc; c2.bnB = bn2sh;
    c2.M = 2 * HW512; c2.N = 64; c2.K = 576; c2.amode = 1; c2.act = 1;
    c2.Cin = 64; c2.Hin = 512; c2.Win = 512; c2.KH = 3; c2.KW = 3;
    c2.stride = 1; c2.pad = 1; c2.Hout = 512; c2.Wout = 512;
    c2.A = e1a; c2.out16 = f1h; gemm(c2);
    c2.A = e1b; c2.out16 = f2h; gemm(c2);

    // ---- patch embed directly on (f1,f2) channel concat (AMODE 2) ----
    GemmP pe{};
    pe.A = f1h; pe.A2 = f2h; pe.Bw = patp; pe.bias = in(15);
    pe.M = LTOK; pe.N = 96; pe.K = 512; pe.amode = 2;
    pe.Cin = 128; pe.Hin = 512; pe.Win = 512; pe.KH = 2; pe.KW = 2;
    pe.stride = 2; pe.pad = 0; pe.Hout = 256; pe.Wout = 256;
    pe.out32 = x_a; gemm(pe);

    // ---- swin blocks ----
    for (int blk = 0; blk < 2; ++blk) {
        const int base = blk ? B1 : B0;
        const int shift = blk ? 4 : 0;
        const _Float16 *qp = blk ? qkvp1 : qkvp0, *pp = blk ? prjp1 : prjp0;
        const _Float16 *f1p = blk ? fc1p1 : fc1p0, *f2p = blk ? fc2p1 : fc2p0;

        ln(x_a, in(base + 0), in(base + 1), tokA);
        GemmP gq{};
        gq.A = tokA; gq.Bw = qp; gq.bias = in(base + 4);
        gq.M = LTOK; gq.N = 288; gq.K = 96; gq.out16 = qkv16;
        gq.nsc = 96; gq.sval = rsqrtf(24.f);   // fold q-scale into epilogue
        gemm(gq);

        hipLaunchKernelGGL(attn_wmma, dim3(2048), dim3(128), 0, stream,
                           qkv16, tokB, in(base + 2), shift);

        GemmP gp{};
        gp.A = tokB; gp.Bw = pp; gp.bias = in(base + 6); gp.resid = x_a;
        gp.M = LTOK; gp.N = 96; gp.K = 96; gp.out32 = x_b;
        gemm(gp);

        ln(x_b, in(base + 7), in(base + 8), tokA);
        GemmP g1{};
        g1.A = tokA; g1.Bw = f1p; g1.bias = in(base + 10); g1.act = 2;
        g1.M = LTOK; g1.N = 384; g1.K = 96; g1.out16 = mlp16;
        gemm(g1);
        GemmP g2{};
        g2.A = mlp16; g2.Bw = f2p; g2.bias = in(base + 12); g2.resid = x_b;
        g2.M = LTOK; g2.N = 96; g2.K = 384; g2.out32 = x_a;
        gemm(g2);
    }

    // ---- final norm + fused conv-transpose (scatter epilogue) ----
    ln(x_a, in(42), in(43), tokA);
    GemmP gt{};
    gt.A = tokA; gt.Bw = ctwp; gt.bias = bt4;
    gt.M = LTOK; gt.N = 256; gt.K = 96; gt.smode = 1;
    gt.out16 = e1a;                              // e1a = gT NHWC [2,512,512,64]
    gemm(gt);

    GemmP gu{};
    gu.A = e1a; gu.Bw = upwp; gu.bias = in(47);
    gu.M = 2 * HW512; gu.N = 64; gu.K = 576; gu.amode = 1; gu.act = 1;
    gu.Cin = 64; gu.Hin = 512; gu.Win = 512; gu.KH = 3; gu.KW = 3;
    gu.stride = 1; gu.pad = 1; gu.Hout = 512; gu.Wout = 512;
    gu.out16 = e1b;                              // e1b = g NHWC
    gemm(gu);

    // ---- mask head directly on (f1,f2,g) channel concat (AMODE 2) ----
    GemmP gm{};
    gm.A = f1h; gm.A2 = f2h; gm.A3 = e1b; gm.Bw = mskp; gm.bias = in(49);
    gm.M = 2 * HW512; gm.N = 64; gm.K = 1728; gm.amode = 2; gm.act = 1;
    gm.Cin = 192; gm.Hin = 512; gm.Win = 512; gm.KH = 3; gm.KW = 3;
    gm.stride = 1; gm.pad = 1; gm.Hout = 512; gm.Wout = 512;
    gm.out16 = e1a;                              // e1a = h NHWC
    gemm(gm);

    {
        const long tot = (long)2 * HW512;
        hipLaunchKernelGGL(final_head, dim3((unsigned)((tot + 255) / 256)), dim3(256),
                           0, stream, e1a, in(50), in(51), img1, img2, (float*)d_out);
    }
}
